// copyBlenderbotAttention_7095285973364
// MI455X (gfx1250) — matrix operand order repaired and checked by reading, not compiled, not run on hardware
//
#include <hip/hip_runtime.h>

// ---------------------------------------------------------------------------
// Blenderbot-style attention layer for MI455X (gfx1250, wave32, WMMA + TDM).
// All matmuls on v_wmma_f32_16x16x32_f16 (fp32 accum). K/V tiles streamed to
// LDS with the Tensor Data Mover (TENSORcnt double buffering), shared by
// 4 waves per block.
// ---------------------------------------------------------------------------

#define EMBED   1024
#define SEQQ    1024
#define BSZN    4
#define NHEADS  16
#define HDIM    64
#define MTOT    (BSZN * SEQQ)          // 4096 rows of X
#define ATTN_CONST_F 3.032639477052158f
#define SCALING_F    0.125f            // 64^-0.5

typedef _Float16 v8h  __attribute__((ext_vector_type(8)));
typedef _Float16 v16h __attribute__((ext_vector_type(16)));
typedef float    v8f  __attribute__((ext_vector_type(8)));
typedef unsigned int u32x4 __attribute__((ext_vector_type(4)));
typedef int          i32x8 __attribute__((ext_vector_type(8)));
typedef int          i32x4 __attribute__((ext_vector_type(4)));

#if __has_builtin(__builtin_amdgcn_tensor_load_to_lds) && \
    __has_builtin(__builtin_amdgcn_s_wait_tensorcnt)
#define HAVE_TDM 1
#else
#define HAVE_TDM 0
#endif

// A-fragment (16x32, 16-bit): lane holds row m = lane%16; halves are
// K = [8g, 8g+8) then [16+8g, 24+8g)  (g = lane/16).  Two b128 loads.
static __device__ __forceinline__ v16h frag_a(const _Float16* row, int g) {
  v8h lo = *(const v8h*)(row + 8 * g);
  v8h hi = *(const v8h*)(row + 16 + 8 * g);
  return __builtin_shufflevector(lo, hi, 0,1,2,3,4,5,6,7,8,9,10,11,12,13,14,15);
}

// B-fragment (32x16, 16-bit): lane holds col n = lane%16; halves are
// K = [8g, 8g+8) then [16+8g, 24+8g)  (g = lane/16).  Two b128 loads.
static __device__ __forceinline__ v16h frag_b(const _Float16* row, int g) {
  v8h lo = *(const v8h*)(row + 8 * g);
  v8h hi = *(const v8h*)(row + 16 + 8 * g);
  return __builtin_shufflevector(lo, hi, 0,1,2,3,4,5,6,7,8,9,10,11,12,13,14,15);
}

static __device__ __forceinline__ v8f wmma16(v16h a, v16h b, v8f c) {
  return __builtin_amdgcn_wmma_f32_16x16x32_f16(false, a, false, b,
                                                (short)0, c, false, false);
}

#if HAVE_TDM
// D# group 0: count=1 | lds_addr | global_addr[56:0] | type=2 ("image")
static __device__ __forceinline__ u32x4 tdm_g0(unsigned lds,
                                               unsigned long long ga) {
  u32x4 g;
  g.x = 1u;                                           // count=1, user mode
  g.y = lds;                                          // LDS byte address
  g.z = (unsigned)ga;                                 // global_addr[31:0]
  g.w = ((unsigned)(ga >> 32) & 0x01FFFFFFu) | (2u << 30);  // [56:32] | type=2
  return g;
}
// D# group 1 for a 2D f16 tile: data_size=2B, tensor dim0/dim1, tile
// dim0/dim1, dim0 stride (all in elements).
static __device__ __forceinline__ i32x8 tdm_g1_2d(unsigned td0, unsigned td1,
                                                  unsigned tile0, unsigned tile1,
                                                  unsigned stride0) {
  i32x8 g = {};
  g[0] = 1 << 16;                                     // data_size = 2 bytes
  g[1] = (int)((td0 & 0xFFFFu) << 16);                // tensor_dim0[15:0]
  g[2] = (int)((td0 >> 16) | ((td1 & 0xFFFFu) << 16));// dim0[31:16]|dim1[15:0]
  g[3] = (int)((td1 >> 16) | (tile0 << 16));          // dim1[31:16]|tile_dim0
  g[4] = (int)(tile1 & 0xFFFFu);                      // tile_dim1 (tile_dim2=0)
  g[5] = (int)stride0;                                // tensor_dim0_stride lo
  return g;                                           // stride hi / dim1 stride = 0
}
static __device__ __forceinline__ void tdm_load(unsigned lds,
                                                const void* gptr,
                                                unsigned td0, unsigned td1,
                                                unsigned tile0, unsigned tile1,
                                                unsigned stride0) {
  i32x4 z4 = {};
  i32x8 z8 = {};
  // 6-arg (clang-23 / therock-10.0) form: g0, g1, g2, g3, g4, cpol
  __builtin_amdgcn_tensor_load_to_lds(
      tdm_g0(lds, (unsigned long long)(size_t)gptr),
      tdm_g1_2d(td0, td1, tile0, tile1, stride0), z4, z4, z8, 0);
}
#endif

// ---------------------------------------------------------------------------
// fp32 -> fp16 cast, 8 elements / thread
// ---------------------------------------------------------------------------
__global__ __launch_bounds__(256)
void cast_f32_to_f16_x8(const float* __restrict__ in,
                        _Float16* __restrict__ out, int n8) {
  int i = blockIdx.x * 256 + threadIdx.x;
  if (i >= n8) return;
  const float4* p = (const float4*)in + (size_t)i * 2;
  float4 a = p[0], b = p[1];
  v8h o;
  o[0]=(_Float16)a.x; o[1]=(_Float16)a.y; o[2]=(_Float16)a.z; o[3]=(_Float16)a.w;
  o[4]=(_Float16)b.x; o[5]=(_Float16)b.y; o[6]=(_Float16)b.z; o[7]=(_Float16)b.w;
  *((v8h*)out + i) = o;
}

// ---------------------------------------------------------------------------
// GEMM: Y = X(4096x1024,f16) @ W(1024x1024,f16)^T, +bias, *scale.
// One wave owns a 64x64 tile: 4 A-frags x 4 B-frags -> 16 WMMAs / K-step
// against 16 b128 loads (1:1 WMMA:VMEM ratio).
// mode 0: f16 (b,h,s,d) | mode 1: f16 (b,h,d,s) | mode 2: f32 row-major
// ---------------------------------------------------------------------------
__global__ __launch_bounds__(256)
void gemm_xwT(const _Float16* __restrict__ X, const _Float16* __restrict__ W,
              const float* __restrict__ bias, float scale,
              _Float16* __restrict__ out16, float* __restrict__ out32,
              int mode) {
  const int lane = threadIdx.x & 31;
  const int nl = lane & 15, g = lane >> 4;
  const int waveId = blockIdx.x * 8 + (threadIdx.x >> 5);  // 1024 waves total
  const int m0 = (waveId >> 4) * 64;                       // 64 row-tiles
  const int n0 = (waveId & 15) * 64;                       // 16 col-tiles

  v8f acc[4][4] = {};

  const _Float16* arow[4];
  const _Float16* brow[4];
#pragma unroll
  for (int i = 0; i < 4; ++i)
    arow[i] = X + (size_t)(m0 + i * 16 + nl) * EMBED;
#pragma unroll
  for (int j = 0; j < 4; ++j)
    brow[j] = W + (size_t)(n0 + j * 16 + nl) * EMBED;

  for (int kk = 0; kk < EMBED; kk += 32) {
    v16h a[4];
#pragma unroll
    for (int i = 0; i < 4; ++i) a[i] = frag_a(arow[i] + kk, g);
#pragma unroll
    for (int j = 0; j < 4; ++j) {
      v16h b = frag_b(brow[j] + kk, g);
#pragma unroll
      for (int i = 0; i < 4; ++i) acc[i][j] = wmma16(a[i], b, acc[i][j]);
    }
  }

#pragma unroll
  for (int i = 0; i < 4; ++i) {
#pragma unroll
    for (int j = 0; j < 4; ++j) {
      const int ncol = n0 + j * 16 + nl;          // C-layout: col = lane%16
      const float bv = bias[ncol];
#pragma unroll
      for (int r = 0; r < 8; ++r) {               // C-layout: row = r + 8g
        const int row = m0 + i * 16 + r + 8 * g;
        const float v = (acc[i][j][r] + bv) * scale;
        if (mode == 2) {
          out32[(size_t)row * EMBED + ncol] = v;
        } else {
          const int b = row >> 10, s = row & 1023;
          const int h = ncol >> 6, d = ncol & 63;
          size_t idx;
          if (mode == 0)
            idx = (((size_t)(b * NHEADS + h) * SEQQ) + s) * HDIM + d;
          else
            idx = (((size_t)(b * NHEADS + h) * HDIM) + d) * SEQQ + s;
          out16[idx] = (_Float16)v;
        }
      }
    }
  }
}

// ---------------------------------------------------------------------------
// Flash attention. Block = 4 waves = 4 consecutive 16-row Q tiles of one
// (b,h); K/V chunks (32 keys) are TDM-DMA'd into double-buffered LDS and
// shared by all 4 waves. Online softmax via LDS + shfl_xor(16).
// ---------------------------------------------------------------------------
#define NITER (SEQQ / 32)

__global__ __launch_bounds__(128)
void flash_attn(const _Float16* __restrict__ Qh, const _Float16* __restrict__ Kh,
                const _Float16* __restrict__ Vt, const float* __restrict__ ids,
                _Float16* __restrict__ AO) {
  __shared__ _Float16 s_k[2][32 * HDIM];     // K chunk: 32 rows x 64 halves
  __shared__ _Float16 s_v[2][HDIM * 32];     // V chunk: 64 d-rows x 32 halves
  __shared__ float    s_sc[4][16 * 32];
  __shared__ _Float16 s_p[4][16 * 32];
  __shared__ float    s_fac[4][16];

  const int tid = threadIdx.x;
  const int lane = tid & 31;
  const int w = tid >> 5;                    // wave 0..3
  const int nl = lane & 15, g = lane >> 4;
  const int qt = blockIdx.x * 4 + w;         // 16-row Q tile (0..63)
  const int bh = blockIdx.y;                 // (batch, head) 0..63

  // Q A-fragments (16x64 tile = two 16x32 frags)
  const _Float16* qbase = Qh + ((size_t)bh * SEQQ + qt * 16 + nl) * HDIM;
  const v16h qa0 = frag_a(qbase, g);
  const v16h qa1 = frag_a(qbase + 32, g);

  v8f o[4] = {};
  float m_i = -3.0e38f, l_i = 0.0f;

  const float* bb = ids + ((size_t)bh * SEQQ + qt * 16) * SEQQ;
  const _Float16* kch = Kh + (size_t)bh * SEQQ * HDIM;   // 32x64 chunks, contig
  const _Float16* vch = Vt + (size_t)bh * HDIM * SEQQ;   // 64 rows, stride 1024

#if HAVE_TDM
  if (tid < 32) {  // wave 0 drives the tensor DMA (EXEC-independent issue)
    tdm_load((unsigned)(size_t)&s_k[0][0], kch, 2048, 1, 2048, 1, 2048);
    tdm_load((unsigned)(size_t)&s_v[0][0], vch, SEQQ, HDIM, 32, HDIM, SEQQ);
  }
#endif

  for (int it = 0; it < NITER; ++it) {
    const int sc = it * 32;
    const int buf = it & 1;

#if HAVE_TDM
    if (tid < 32) {
      if (it + 1 < NITER) {   // prefetch next chunk into the other buffer
        tdm_load((unsigned)(size_t)&s_k[buf ^ 1][0],
                 kch + (size_t)(sc + 32) * HDIM, 2048, 1, 2048, 1, 2048);
        tdm_load((unsigned)(size_t)&s_v[buf ^ 1][0],
                 vch + sc + 32, SEQQ, HDIM, 32, HDIM, SEQQ);
        __builtin_amdgcn_s_wait_tensorcnt(2);  // current chunk complete
      } else {
        __builtin_amdgcn_s_wait_tensorcnt(0);
      }
    }
#else
    {  // cooperative fallback copy (128 threads)
      v8h* dk = (v8h*)&s_k[buf][0];
      const v8h* sk = (const v8h*)(kch + (size_t)sc * HDIM);
      dk[tid * 2] = sk[tid * 2];
      dk[tid * 2 + 1] = sk[tid * 2 + 1];
      const int row = tid >> 1, hh = (tid & 1) * 16;
      v8h* dv = (v8h*)&s_v[buf][row * 32 + hh];
      const v8h* sv = (const v8h*)(vch + (size_t)row * SEQQ + sc + hh);
      dv[0] = sv[0];
      dv[1] = sv[1];
    }
#endif
    __syncthreads();

    // ---- two 16x16 score tiles: S = Q(16x64) x K_chunk^T, + bias ----
#pragma unroll
    for (int j = 0; j < 2; ++j) {
      const _Float16* kb = &s_k[buf][(j * 16 + nl) * HDIM];
      v16h kb0 = frag_b(kb, g);
      v16h kb1 = frag_b(kb + 32, g);
      v8f s = {};
      s = wmma16(qa0, kb0, s);
      s = wmma16(qa1, kb1, s);
#pragma unroll
      for (int r = 0; r < 8; ++r) {
        const int row = r + 8 * g;
        const float* bp = bb + (size_t)row * SEQQ + sc + j * 16 + nl;
        __builtin_prefetch(bp + 32, 0, 0);      // next chunk's bias stream
        const float v = s[r] + ATTN_CONST_F * bp[0];
        s_sc[w][row * 32 + j * 16 + nl] = v;
      }
    }

    // ---- online softmax: lane handles row nl, 16 cols starting g*16 ----
    float vbuf[16];
    float mloc = -3.0e38f;
#pragma unroll
    for (int c = 0; c < 16; ++c) {
      vbuf[c] = s_sc[w][nl * 32 + g * 16 + c];
      mloc = fmaxf(mloc, vbuf[c]);
    }
    mloc = fmaxf(mloc, __shfl_xor(mloc, 16));
    const float mnew = fmaxf(m_i, mloc);
    const float fac = __expf(m_i - mnew);
    float ls = 0.0f;
#pragma unroll
    for (int c = 0; c < 16; ++c) {
      const float p = __expf(vbuf[c] - mnew);
      ls += p;
      s_p[w][nl * 32 + g * 16 + c] = (_Float16)p;
    }
    ls += __shfl_xor(ls, 16);
    l_i = l_i * fac + ls;
    m_i = mnew;
    if (g == 0) s_fac[w][nl] = fac;

    // ---- rescale O accumulators by per-row factor ----
    float fr[8];
#pragma unroll
    for (int r = 0; r < 8; ++r) fr[r] = s_fac[w][r + 8 * g];
#pragma unroll
    for (int t = 0; t < 4; ++t)
#pragma unroll
      for (int r = 0; r < 8; ++r) o[t][r] *= fr[r];

    // ---- P A-frag (16x32 f16 from LDS) x V chunk (32x64 from LDS) ----
    const v16h pa = frag_a(&s_p[w][nl * 32], g);
#pragma unroll
    for (int t = 0; t < 4; ++t) {
      v16h vb = frag_b(&s_v[buf][(t * 16 + nl) * 32], g);
      o[t] = wmma16(pa, vb, o[t]);
    }
    __syncthreads();   // all waves done with buf before DMA overwrites it
  }

  // ---- normalize by row sum and store as (b, s, e) f16 ----
  if (g == 0) s_fac[w][nl] = 1.0f / l_i;
  __syncthreads();
  float fr[8];
#pragma unroll
  for (int r = 0; r < 8; ++r) fr[r] = s_fac[w][r + 8 * g];
  const int b = bh >> 4, h = bh & 15;
#pragma unroll
  for (int t = 0; t < 4; ++t)
#pragma unroll
    for (int r = 0; r < 8; ++r) {
      const int srow = qt * 16 + r + 8 * g;
      const size_t idx =
          ((size_t)b * SEQQ + srow) * EMBED + h * 64 + t * 16 + nl;
      AO[idx] = (_Float16)(o[t][r] * fr[r]);
    }
}

// ---------------------------------------------------------------------------
// Launcher.  Workspace (48 MiB):
//   [ 0M) Xh f16  [ 8M) Wq  [10M) Wk  [12M) Wv  [14M) Wo  (f16, 2M each)
//   [16M) Qh  [24M) Kh  [32M) Vt  [40M) AO     (f16, 8M each)
// ---------------------------------------------------------------------------
extern "C" void kernel_launch(void* const* d_in, const int* in_sizes, int n_in,
                              void* d_out, int out_size, void* d_ws,
                              size_t ws_size, hipStream_t stream) {
  (void)in_sizes; (void)n_in; (void)out_size; (void)ws_size;
  const float* hidden = (const float*)d_in[0];
  const float* ids    = (const float*)d_in[1];
  const float* Wq = (const float*)d_in[2]; const float* bq = (const float*)d_in[3];
  const float* Wk = (const float*)d_in[4]; const float* bk = (const float*)d_in[5];
  const float* Wv = (const float*)d_in[6]; const float* bv = (const float*)d_in[7];
  const float* Wo = (const float*)d_in[8]; const float* bo = (const float*)d_in[9];

  char* ws = (char*)d_ws;
  const size_t MB = 1u << 20;
  _Float16* Xh  = (_Float16*)(ws + 0 * MB);
  _Float16* Wqh = (_Float16*)(ws + 8 * MB);
  _Float16* Wkh = (_Float16*)(ws + 10 * MB);
  _Float16* Wvh = (_Float16*)(ws + 12 * MB);
  _Float16* Woh = (_Float16*)(ws + 14 * MB);
  _Float16* Qh  = (_Float16*)(ws + 16 * MB);
  _Float16* Kh  = (_Float16*)(ws + 24 * MB);
  _Float16* Vt  = (_Float16*)(ws + 32 * MB);
  _Float16* AO  = (_Float16*)(ws + 40 * MB);

  // 1) fp32 -> fp16 casts
  {
    int n8 = (MTOT * EMBED) / 8;
    cast_f32_to_f16_x8<<<(n8 + 255) / 256, 256, 0, stream>>>(hidden, Xh, n8);
    int w8 = (EMBED * EMBED) / 8;
    cast_f32_to_f16_x8<<<(w8 + 255) / 256, 256, 0, stream>>>(Wq, Wqh, w8);
    cast_f32_to_f16_x8<<<(w8 + 255) / 256, 256, 0, stream>>>(Wk, Wkh, w8);
    cast_f32_to_f16_x8<<<(w8 + 255) / 256, 256, 0, stream>>>(Wv, Wvh, w8);
    cast_f32_to_f16_x8<<<(w8 + 255) / 256, 256, 0, stream>>>(Wo, Woh, w8);
  }

  // 2) Q/K/V projections (1024 waves of 64x64 tiles -> 128 blocks x 256 thr)
  gemm_xwT<<<128, 256, 0, stream>>>(Xh, Wqh, bq, SCALING_F, Qh, nullptr, 0);
  gemm_xwT<<<128, 256, 0, stream>>>(Xh, Wkh, bk, 1.0f,      Kh, nullptr, 0);
  gemm_xwT<<<128, 256, 0, stream>>>(Xh, Wvh, bv, 1.0f,      Vt, nullptr, 1);

  // 3) attention: 16 q-groups x 64 (b,h); 4 waves/block share TDM'd K/V
  flash_attn<<<dim3(16, 64), 128, 0, stream>>>(Qh, Kh, Vt, ids, AO);

  // 4) output projection -> fp32 d_out
  gemm_xwT<<<128, 256, 0, stream>>>(AO, Woh, bo, 1.0f, nullptr, (float*)d_out, 2);
}
